// DeepseekV4Attention_7610682048940
// MI455X (gfx1250) — matrix operand order repaired and checked by reading, not compiled, not run on hardware
//
#include <hip/hip_runtime.h>
#include <hip/hip_bf16.h>
#include <math.h>

// ---------------- problem constants ----------------
#define S_    2048
#define HID_  2048
#define H_    16
#define D_    256
#define R_    64
#define NOPE_ 192
#define QL_   1024
#define OR_   512
#define G_    4
#define EPS_  1e-6f
#define SCALE_ 0.0625f   // 256^-0.5

typedef __attribute__((ext_vector_type(16))) __bf16 v16bf;
typedef __attribute__((ext_vector_type(8)))  float  v8f;
typedef __attribute__((ext_vector_type(4)))  unsigned int v4u;
typedef __attribute__((ext_vector_type(8)))  int v8i;
typedef __attribute__((ext_vector_type(4)))  int v4i;

// ---------------- helpers ----------------
__device__ __forceinline__ unsigned short f2bf(float x) {
  unsigned int u = __float_as_uint(x);
  unsigned int r = u + 0x7FFFu + ((u >> 16) & 1u);   // round-to-nearest-even
  return (unsigned short)(r >> 16);
}

// A-operand fragment: 16x32 (MxK), row-major source [rows x K].
// Lanes 0-15 row M=lane, K={0..7,16..23}; lanes 16-31 same rows, K={8..15,24..31}.
__device__ __forceinline__ v16bf load_fragA(const unsigned short* __restrict__ base,
                                            int ld, int row0, int k0) {
  const int lane = threadIdx.x & 31;
  const int r  = lane & 15;
  const int hi = lane >> 4;
  const unsigned short* p = base + (size_t)(row0 + r) * (size_t)ld + k0 + hi * 8;
  union { uint4 u[2]; v16bf v; } f;
  f.u[0] = *reinterpret_cast<const uint4*>(p);
  f.u[1] = *reinterpret_cast<const uint4*>(p + 16);
  return f.v;
}

// B-operand fragment: 32x16 (KxN) from row-major W[N x K] (Wt form).
// Lanes 0-15 column N=lane, K={0..7,16..23}; lanes 16-31 K={8..15,24..31}.
__device__ __forceinline__ v16bf load_fragB(const unsigned short* __restrict__ base,
                                            int ld, int n0, int k0) {
  const int lane = threadIdx.x & 31;
  const int r  = lane & 15;
  const int hi = lane >> 4;
  const unsigned short* p = base + (size_t)(n0 + r) * (size_t)ld + k0 + hi * 8;
  union { uint4 u[2]; v16bf v; } f;
  f.u[0] = *reinterpret_cast<const uint4*>(p);
  f.u[1] = *reinterpret_cast<const uint4*>(p + 16);
  return f.v;
}

__device__ __forceinline__ v8f wmma_bf16(v16bf a, v16bf b, v8f c) {
  return __builtin_amdgcn_wmma_f32_16x16x32_bf16(false, a, false, b, (short)0, c,
                                                 false, false);
}

// ---- Tensor Data Mover: 2-D tile load Global->LDS (D# per ISA cdna5 §8) ----
// data_size = 2 bytes.  tile_d0 = contiguous elements per row, tile_d1 = rows,
// stride0 = row pitch in elements.  Groups 2/3 zero (2-D tensor).
// This toolchain's builtin is the 6-arg form: (v4u, v8i, v4i, v4i, v8i, i32).
__device__ __forceinline__ void tdm_load_2d(unsigned int lds_addr,
                                            const unsigned short* gptr,
                                            unsigned int tensor_d0,
                                            unsigned int tensor_d1,
                                            unsigned int tile_d0,
                                            unsigned int tile_d1,
                                            unsigned int stride0) {
  unsigned long long ga = (unsigned long long)(size_t)gptr;
  v4u g0;
  g0.x = 1u;                                             // count=1, user mode
  g0.y = lds_addr;                                       // lds_addr [63:32]
  g0.z = (unsigned int)(ga & 0xffffffffull);             // global_addr[31:0]
  g0.w = (unsigned int)((ga >> 32) & 0x01ffffffull) | 0x80000000u; // type=2
  v8i g1;
  g1[0] = 0x00010000;                                    // data_size=2B
  g1[1] = (int)((tensor_d0 & 0xffffu) << 16);            // tensor_dim0 lo16
  g1[2] = (int)(((tensor_d0 >> 16) & 0xffffu) |
                ((tensor_d1 & 0xffffu) << 16));          // dim0 hi / dim1 lo
  g1[3] = (int)(((tensor_d1 >> 16) & 0xffffu) |
                ((tile_d0 & 0xffffu) << 16));            // dim1 hi / tile_dim0
  g1[4] = (int)(tile_d1 & 0xffffu);                      // tile_dim1, tile_dim2=0
  g1[5] = (int)stride0;                                  // tensor_dim0_stride lo32
  g1[6] = 0;                                             // stride0 hi / stride1 lo
  g1[7] = 0;
  v4i gz4 = {0, 0, 0, 0};
  v8i gz8 = {0, 0, 0, 0, 0, 0, 0, 0};
  __builtin_amdgcn_tensor_load_to_lds(g0, g1, gz4, gz4, gz8, 0);
}

__device__ __forceinline__ float block_sum256(float v) {
  __shared__ float red[256];
  const int t = threadIdx.x;
  red[t] = v;
  __syncthreads();
  #pragma unroll
  for (int o = 128; o > 0; o >>= 1) {
    if (t < o) red[t] += red[t + o];
    __syncthreads();
  }
  float r = red[0];
  __syncthreads();
  return r;
}

// ---------------- elementwise kernels ----------------
__global__ void k_f32_to_bf16(const float* __restrict__ in,
                              unsigned short* __restrict__ out, int n) {
  int i = blockIdx.x * blockDim.x + threadIdx.x;
  int stride = gridDim.x * blockDim.x;
  for (; i < n; i += stride) out[i] = f2bf(in[i]);
}

__global__ void k_cossin(const float* __restrict__ f, float* __restrict__ c,
                         float* __restrict__ s, int n) {
  int i = blockIdx.x * blockDim.x + threadIdx.x;
  if (i < n) { c[i] = __cosf(f[i]); s[i] = __sinf(f[i]); }
}

// ---------------- generic bf16 GEMM: C[M,N] = A[M,K] * W[N,K]^T ----------------
__global__ __launch_bounds__(256)
void k_gemm_bf16(const unsigned short* __restrict__ A,
                 const unsigned short* __restrict__ W,
                 float* __restrict__ C,
                 int K, int lda, int ldw, int ldc,
                 int groupN, int groupStrideA) {
  const int wid   = threadIdx.x >> 5;
  const int lane  = threadIdx.x & 31;
  const int waveM = wid >> 2;           // 0..1
  const int waveN = wid & 3;            // 0..3
  const int rowBase = blockIdx.y * 64 + waveM * 32;
  const int colBase = blockIdx.x * 256 + waveN * 64;
  const int aOff    = (colBase / groupN) * groupStrideA;
  const unsigned short* Ab = A + aOff;

  v8f zero = {};
  v8f acc[2][4];
  #pragma unroll
  for (int i = 0; i < 2; ++i)
    #pragma unroll
    for (int j = 0; j < 4; ++j) acc[i][j] = zero;

  for (int k0 = 0; k0 < K; k0 += 32) {
    if (k0 + 32 < K) {
      __builtin_prefetch(Ab + (size_t)rowBase * lda + k0 + 32, 0, 1);
      __builtin_prefetch(W  + (size_t)colBase * ldw + k0 + 32, 0, 1);
    }
    v16bf af[2];
    af[0] = load_fragA(Ab, lda, rowBase,      k0);
    af[1] = load_fragA(Ab, lda, rowBase + 16, k0);
    v16bf wf[4];
    #pragma unroll
    for (int j = 0; j < 4; ++j)
      wf[j] = load_fragB(W, ldw, colBase + 16 * j, k0);
    #pragma unroll
    for (int i = 0; i < 2; ++i)
      #pragma unroll
      for (int j = 0; j < 4; ++j)
        acc[i][j] = wmma_bf16(af[i], wf[j], acc[i][j]);
  }

  const int r  = lane & 15;
  const int hi = lane >> 4;
  #pragma unroll
  for (int i = 0; i < 2; ++i)
    #pragma unroll
    for (int j = 0; j < 4; ++j)
      #pragma unroll
      for (int v = 0; v < 8; ++v)
        C[(size_t)(rowBase + 16 * i + v + 8 * hi) * ldc + colBase + 16 * j + r] =
            acc[i][j][v];
}

// ---------------- RMS norm over QL=1024 with weight -> bf16 ----------------
__global__ __launch_bounds__(256)
void k_rms_qlat(const float* __restrict__ in, const float* __restrict__ w,
                unsigned short* __restrict__ out) {
  const int s = blockIdx.x;
  const int t = threadIdx.x;
  float v[4], ss = 0.f;
  #pragma unroll
  for (int i = 0; i < 4; ++i) {
    v[i] = in[(size_t)s * QL_ + t + 256 * i];
    ss += v[i] * v[i];
  }
  float tot = block_sum256(ss);
  float rinv = rsqrtf(tot * (1.0f / QL_) + EPS_);
  #pragma unroll
  for (int i = 0; i < 4; ++i)
    out[(size_t)s * QL_ + t + 256 * i] = f2bf(v[i] * rinv * w[t + 256 * i]);
}

// ---------------- per-head q norm (no weight) + RoPE -> bf16 ----------------
__global__ __launch_bounds__(256)
void k_qnorm_rope(const float* __restrict__ q, const float* __restrict__ cosb,
                  const float* __restrict__ sinb, unsigned short* __restrict__ out) {
  const int h = blockIdx.x, s = blockIdx.y, d = threadIdx.x;
  const size_t idx = (size_t)s * (H_ * D_) + h * D_ + d;
  float v = q[idx];
  float tot = block_sum256(v * v);
  v *= rsqrtf(tot * (1.0f / D_) + EPS_);
  __shared__ float pe[R_];
  if (d >= NOPE_) pe[d - NOPE_] = v;
  __syncthreads();
  if (d >= NOPE_) {
    int j = (d - NOPE_) >> 1;
    float c = cosb[s * (R_ / 2) + j], sn = sinb[s * (R_ / 2) + j];
    float x1 = pe[(d - NOPE_) & ~1], x2 = pe[(d - NOPE_) | 1];
    v = ((d & 1) == 0) ? (x1 * c - x2 * sn) : (x1 * sn + x2 * c);
  }
  out[idx] = f2bf(v);
}

// ---------------- kv: RMS (weighted) + RoPE -> kv_bf and kvT_bf ----------------
__global__ __launch_bounds__(256)
void k_kv_prep(const float* __restrict__ kvraw, const float* __restrict__ w,
               const float* __restrict__ cosb, const float* __restrict__ sinb,
               unsigned short* __restrict__ kv_bf, unsigned short* __restrict__ kvT_bf) {
  const int s = blockIdx.x, d = threadIdx.x;
  float v = kvraw[(size_t)s * D_ + d];
  float tot = block_sum256(v * v);
  v = v * rsqrtf(tot * (1.0f / D_) + EPS_) * w[d];
  __shared__ float pe[R_];
  if (d >= NOPE_) pe[d - NOPE_] = v;
  __syncthreads();
  if (d >= NOPE_) {
    int j = (d - NOPE_) >> 1;
    float c = cosb[s * (R_ / 2) + j], sn = sinb[s * (R_ / 2) + j];
    float x1 = pe[(d - NOPE_) & ~1], x2 = pe[(d - NOPE_) | 1];
    v = ((d & 1) == 0) ? (x1 * c - x2 * sn) : (x1 * sn + x2 * c);
  }
  unsigned short b = f2bf(v);
  kv_bf[(size_t)s * D_ + d]  = b;
  kvT_bf[(size_t)d * S_ + s] = b;
}

// ---------------- flash attention with TDM double-buffered KV tiles ----------
// Every wave in the block shares the same 32-key tile -> wave 0 TDM-loads it
// into LDS (kvA: 32x256 for QK^T, kvT: 256x32 for PV) while the block computes
// the previous tile; s_wait_tensorcnt + s_barrier flips the double buffer.
__global__ __launch_bounds__(256)
void k_attn(const unsigned short* __restrict__ q_bf,    // S x (H*D)
            const unsigned short* __restrict__ kv_bf,   // S x D
            const unsigned short* __restrict__ kvT_bf,  // D x S
            const float* __restrict__ attn_sink,
            float* __restrict__ o_f) {                  // S x (H*D)
  __shared__ unsigned short kvA[2][32 * D_];   // [t-local][d]
  __shared__ unsigned short kvT[2][D_ * 32];   // [d][t-local]

  const int h    = blockIdx.y;
  const int wid  = threadIdx.x >> 5;
  const int lane = threadIdx.x & 31;
  const int q0   = blockIdx.x * 128 + wid * 16;
  const int r    = lane & 15;
  const int hi   = lane >> 4;
  const int sQ   = q0 + r;
  const int qhi  = q0 + 15;
  const int ntiles = blockIdx.x * 4 + 4;       // block-uniform causal tile count

  v16bf qf[8];
  #pragma unroll
  for (int kk = 0; kk < 8; ++kk)
    qf[kk] = load_fragB(q_bf + h * D_, H_ * D_, q0, kk * 32);

  v8f zero = {};
  v8f acc[16];
  #pragma unroll
  for (int dt = 0; dt < 16; ++dt) acc[dt] = zero;
  float m = -1e30f, l = 0.f;

  // prologue: stage tile 0
  if (wid == 0) {
    tdm_load_2d((unsigned int)(size_t)&kvA[0][0], kv_bf,
                D_, S_, D_, 32, D_);
    tdm_load_2d((unsigned int)(size_t)&kvT[0][0], kvT_bf,
                S_, D_, 32, D_, S_);
    __builtin_amdgcn_s_wait_tensorcnt(0);
  }
  __syncthreads();

  for (int it = 0; it < ntiles; ++it) {
    const int t0  = it * 32;
    const int cur = it & 1;
    if (wid == 0 && it + 1 < ntiles) {        // async prefetch of next tile
      const int tn = t0 + 32;
      tdm_load_2d((unsigned int)(size_t)&kvA[cur ^ 1][0],
                  kv_bf + (size_t)tn * D_, D_, S_, D_, 32, D_);
      tdm_load_2d((unsigned int)(size_t)&kvT[cur ^ 1][0],
                  kvT_bf + tn, S_, D_, 32, D_, S_);
    }

    if (t0 <= qhi) {                          // wave-uniform causal skip
      v8f c1 = zero, c2 = zero;
      #pragma unroll
      for (int kk = 0; kk < 8; ++kk) {
        union { uint4 u[2]; v16bf v; } a1, a2;
        const int ka = kk * 32 + hi * 8;
        a1.u[0] = *reinterpret_cast<const uint4*>(&kvA[cur][r * D_ + ka]);
        a1.u[1] = *reinterpret_cast<const uint4*>(&kvA[cur][r * D_ + ka + 16]);
        a2.u[0] = *reinterpret_cast<const uint4*>(&kvA[cur][(r + 16) * D_ + ka]);
        a2.u[1] = *reinterpret_cast<const uint4*>(&kvA[cur][(r + 16) * D_ + ka + 16]);
        c1 = wmma_bf16(a1.v, qf[kk], c1);
        c2 = wmma_bf16(a2.v, qf[kk], c2);
      }
      // causal mask + scale + running max
      float sc1[8], sc2[8];
      float tmax = -1e30f;
      #pragma unroll
      for (int v = 0; v < 8; ++v) {
        int t1 = t0 + hi * 8 + v;
        int t2 = t1 + 16;
        sc1[v] = (t1 <= sQ) ? c1[v] * SCALE_ : -1e30f;
        sc2[v] = (t2 <= sQ) ? c2[v] * SCALE_ : -1e30f;
        tmax = fmaxf(tmax, fmaxf(sc1[v], sc2[v]));
      }
      tmax = fmaxf(tmax, __shfl_xor(tmax, 16, 32));
      float nm = fmaxf(m, tmax);
      float psum = 0.f;
      union { unsigned short u[16]; v16bf vec; } pf;  // P A-fragment, in-lane
      #pragma unroll
      for (int v = 0; v < 8; ++v) {
        float p1 = __expf(sc1[v] - nm);
        float p2 = __expf(sc2[v] - nm);
        psum += p1 + p2;
        pf.u[v]     = f2bf(p1);
        pf.u[v + 8] = f2bf(p2);
      }
      psum += __shfl_xor(psum, 16, 32);
      float esc = __expf(m - nm);
      l = l * esc + psum;
      m = nm;
      float ev[8];
      #pragma unroll
      for (int v = 0; v < 8; ++v) ev[v] = __shfl(esc, v + 8 * hi, 32);
      #pragma unroll
      for (int dt = 0; dt < 16; ++dt) {
        #pragma unroll
        for (int v = 0; v < 8; ++v) acc[dt][v] *= ev[v];
        union { uint4 u[2]; v16bf v; } bf;
        const unsigned short* p = &kvT[cur][(dt * 16 + r) * 32 + hi * 8];
        bf.u[0] = *reinterpret_cast<const uint4*>(p);
        bf.u[1] = *reinterpret_cast<const uint4*>(p + 16);
        acc[dt] = wmma_bf16(pf.vec, bf.v, acc[dt]);
      }
    }

    if (wid == 0 && it + 1 < ntiles) __builtin_amdgcn_s_wait_tensorcnt(0);
    __syncthreads();
  }

  // sink logit + final normalization
  float sink = attn_sink[h];
  float nm = fmaxf(m, sink);
  float denom = l * __expf(m - nm) + __expf(sink - nm);
  float fs = __expf(m - nm) / denom;
  float fv[8];
  #pragma unroll
  for (int v = 0; v < 8; ++v) fv[v] = __shfl(fs, v + 8 * hi, 32);
  #pragma unroll
  for (int dt = 0; dt < 16; ++dt)
    #pragma unroll
    for (int v = 0; v < 8; ++v)
      o_f[(size_t)(q0 + v + 8 * hi) * (H_ * D_) + h * D_ + dt * 16 + r] =
          acc[dt][v] * fv[v];
}

// ---------------- inverse RoPE on o_pe -> bf16 ----------------
__global__ __launch_bounds__(256)
void k_o_post(const float* __restrict__ o, const float* __restrict__ cosb,
              const float* __restrict__ sinb, unsigned short* __restrict__ out) {
  const int h = blockIdx.x, s = blockIdx.y, d = threadIdx.x;
  const size_t idx = (size_t)s * (H_ * D_) + h * D_ + d;
  float v = o[idx];
  __shared__ float pe[R_];
  if (d >= NOPE_) pe[d - NOPE_] = v;
  __syncthreads();
  if (d >= NOPE_) {
    int j = (d - NOPE_) >> 1;
    float c = cosb[s * (R_ / 2) + j], sn = -sinb[s * (R_ / 2) + j];  // conj
    float x1 = pe[(d - NOPE_) & ~1], x2 = pe[(d - NOPE_) | 1];
    v = ((d & 1) == 0) ? (x1 * c - x2 * sn) : (x1 * sn + x2 * c);
  }
  out[idx] = f2bf(v);
}

// ---------------- host orchestration ----------------
extern "C" void kernel_launch(void* const* d_in, const int* in_sizes, int n_in,
                              void* d_out, int out_size, void* d_ws, size_t ws_size,
                              hipStream_t stream) {
  const float* x        = (const float*)d_in[0];
  const float* freqs    = (const float*)d_in[1];
  const float* wq_a     = (const float*)d_in[2];
  const float* q_norm_w = (const float*)d_in[3];
  const float* wq_b     = (const float*)d_in[4];
  const float* wkv      = (const float*)d_in[5];
  const float* kv_norm_w= (const float*)d_in[6];
  const float* wo_a     = (const float*)d_in[7];
  const float* wo_b     = (const float*)d_in[8];
  const float* sink     = (const float*)d_in[9];
  float* out = (float*)d_out;

  char* base = (char*)d_ws;
  size_t off = 0;
  auto alloc = [&](size_t bytes) -> void* {
    void* p = base + off;
    off = (off + bytes + 255) & ~(size_t)255;
    return p;
  };
  typedef unsigned short bfu;
  bfu*   x_bf     = (bfu*)alloc((size_t)S_ * HID_ * 2);
  bfu*   wqa_bf   = (bfu*)alloc((size_t)QL_ * HID_ * 2);
  bfu*   wqb_bf   = (bfu*)alloc((size_t)H_ * D_ * QL_ * 2);
  bfu*   wkv_bf   = (bfu*)alloc((size_t)D_ * HID_ * 2);
  bfu*   woa_bf   = (bfu*)alloc((size_t)G_ * OR_ * 1024 * 2);
  bfu*   wob_bf   = (bfu*)alloc((size_t)HID_ * G_ * OR_ * 2);
  float* cosb     = (float*)alloc((size_t)S_ * (R_ / 2) * 4);
  float* sinb     = (float*)alloc((size_t)S_ * (R_ / 2) * 4);
  float* qlat_f   = (float*)alloc((size_t)S_ * QL_ * 4);
  bfu*   qlat_bf  = (bfu*)alloc((size_t)S_ * QL_ * 2);
  float* q_f      = (float*)alloc((size_t)S_ * H_ * D_ * 4);  // reused as o_f
  bfu*   q_bf     = (bfu*)alloc((size_t)S_ * H_ * D_ * 2);
  float* kvraw_f  = (float*)alloc((size_t)S_ * D_ * 4);
  bfu*   kv_bf    = (bfu*)alloc((size_t)S_ * D_ * 2);
  bfu*   kvT_bf   = (bfu*)alloc((size_t)D_ * S_ * 2);
  bfu*   o_bf     = (bfu*)alloc((size_t)S_ * H_ * D_ * 2);
  float* out2_f   = (float*)alloc((size_t)S_ * G_ * OR_ * 4);
  bfu*   out2_bf  = (bfu*)alloc((size_t)S_ * G_ * OR_ * 2);
  float* o_f      = q_f;

  auto cvt = [&](const float* in, bfu* o, size_t n) {
    int blocks = (int)((n + 1023) / 1024);
    if (blocks > 4096) blocks = 4096;
    k_f32_to_bf16<<<blocks, 256, 0, stream>>>(in, o, (int)n);
  };
  cvt(x,    x_bf,   (size_t)S_ * HID_);
  cvt(wq_a, wqa_bf, (size_t)QL_ * HID_);
  cvt(wq_b, wqb_bf, (size_t)H_ * D_ * QL_);
  cvt(wkv,  wkv_bf, (size_t)D_ * HID_);
  cvt(wo_a, woa_bf, (size_t)G_ * OR_ * 1024);
  cvt(wo_b, wob_bf, (size_t)HID_ * G_ * OR_);
  k_cossin<<<(S_ * (R_ / 2) + 255) / 256, 256, 0, stream>>>(
      freqs, cosb, sinb, S_ * (R_ / 2));

  k_gemm_bf16<<<dim3(QL_ / 256, S_ / 64), 256, 0, stream>>>(
      x_bf, wqa_bf, qlat_f, HID_, HID_, HID_, QL_, QL_, 0);
  k_rms_qlat<<<S_, 256, 0, stream>>>(qlat_f, q_norm_w, qlat_bf);

  k_gemm_bf16<<<dim3(H_ * D_ / 256, S_ / 64), 256, 0, stream>>>(
      qlat_bf, wqb_bf, q_f, QL_, QL_, QL_, H_ * D_, H_ * D_, 0);
  k_qnorm_rope<<<dim3(H_, S_), 256, 0, stream>>>(q_f, cosb, sinb, q_bf);

  k_gemm_bf16<<<dim3(D_ / 256, S_ / 64), 256, 0, stream>>>(
      x_bf, wkv_bf, kvraw_f, HID_, HID_, HID_, D_, D_, 0);
  k_kv_prep<<<S_, 256, 0, stream>>>(kvraw_f, kv_norm_w, cosb, sinb, kv_bf, kvT_bf);

  k_attn<<<dim3(S_ / 128, H_), 256, 0, stream>>>(q_bf, kv_bf, kvT_bf, sink, o_f);
  k_o_post<<<dim3(H_, S_), 256, 0, stream>>>(o_f, cosb, sinb, o_bf);

  k_gemm_bf16<<<dim3(G_ * OR_ / 256, S_ / 64), 256, 0, stream>>>(
      o_bf, woa_bf, out2_f, 1024, H_ * D_, 1024, G_ * OR_, OR_, 1024);
  cvt(out2_f, out2_bf, (size_t)S_ * G_ * OR_);

  k_gemm_bf16<<<dim3(HID_ / 256, S_ / 64), 256, 0, stream>>>(
      out2_bf, wob_bf, out, G_ * OR_, G_ * OR_, G_ * OR_, HID_, HID_, 0);

  (void)in_sizes; (void)n_in; (void)out_size; (void)ws_size;
}
